// minTGU_ME_50697793962783
// MI455X (gfx1250) — compile-verified, matrix operand order checked by reading, not run on hardware
//
#include <hip/hip_runtime.h>
#include <hip/hip_bf16.h>

typedef __bf16 bf16_t;
typedef __attribute__((ext_vector_type(8)))  bf16_t v8bf;
typedef __attribute__((ext_vector_type(16))) bf16_t v16bf;
typedef __attribute__((ext_vector_type(8)))  float  v8f;
typedef __attribute__((ext_vector_type(4)))  int    v4i;

union Frag { v16bf v; v8bf h[2]; };

#if __has_builtin(__builtin_amdgcn_global_load_async_to_lds_b128)
#define HAVE_ASYNC_LDS 1
#else
#define HAVE_ASYNC_LDS 0
#endif

__device__ __forceinline__ void wait_async0() {
#if __has_builtin(__builtin_amdgcn_s_wait_asynccnt)
    __builtin_amdgcn_s_wait_asynccnt(0);
#else
    asm volatile("s_wait_asynccnt 0" ::: "memory");
#endif
}

// ---------------------------------------------------------------------------
// math helpers (fast hardware transcendentals)
// ---------------------------------------------------------------------------
__device__ __forceinline__ float log_sigmoid(float x) {
    return fminf(x, 0.0f) - log1pf(__expf(-fabsf(x)));
}
__device__ __forceinline__ float log_g(float x) {
    return (x >= 0.0f) ? __logf(x + 0.5f) : (x - log1pf(__expf(x)));
}
__device__ __forceinline__ float logaddexp(float a, float b) {
    float mx = fmaxf(a, b), mn = fminf(a, b);
    return mx + log1pf(__expf(mn - mx));
}

// ---------------------------------------------------------------------------
// GEMM 1: G[M, 5E] = X[M,K] * W[rowmap(n), K]^T + bias   (K = 1024)
//   Block tile 256M x 64N, 8 waves, wave tile 32x64 -> 8 WMMA / K-step.
//   Software-pipelined: next K-tile loads issued during WMMA compute.
// ---------------------------------------------------------------------------
__global__ __launch_bounds__(256)
void gemm_gates_kernel(const float* __restrict__ X,
                       const float* __restrict__ W,
                       const float* __restrict__ bias,
                       float* __restrict__ G,
                       int M, int K, int E)
{
    const int N5 = 5 * E;
    const int bm = blockIdx.x * 256;
    const int bn = blockIdx.y * 64;
    const int wRowBase = bn + (bn >= 4 * E ? E : 0);

    const int tid  = threadIdx.x;
    const int lane = tid & 31;
    const int wave = tid >> 5;        // 0..7 -> 32-row strip
    const int hsel = lane >> 4;
    const int l16  = lane & 15;

    __shared__ v8bf As[256][6];       // 4 chunks used + 2 pad (bank stagger)
    __shared__ v8bf Bs[64][6];
    bf16_t* AsS = (bf16_t*)&As[0][0]; // row stride 48 shorts
    bf16_t* BsS = (bf16_t*)&Bs[0][0];

    v8f acc[2][4] = {};

    float4 ra[8], rb[2];
    // ---- preload K-tile 0 into registers ----
    #pragma unroll
    for (int i = 0; i < 8; ++i) {
        int f = tid + 256 * i, row = f >> 3, c4 = f & 7;
        ra[i] = *(const float4*)(X + (size_t)(bm + row) * K + c4 * 4);
    }
    #pragma unroll
    for (int i = 0; i < 2; ++i) {
        int f = tid + 256 * i, row = f >> 3, c4 = f & 7;
        rb[i] = *(const float4*)(W + (size_t)(wRowBase + row) * K + c4 * 4);
    }

    for (int k0 = 0; k0 < K; k0 += 32) {
        // ---- convert staged registers -> LDS ----
        #pragma unroll
        for (int i = 0; i < 8; ++i) {
            int f = tid + 256 * i, row = f >> 3, c4 = f & 7;
            bf16_t* d = AsS + row * 48 + c4 * 4;
            d[0] = (bf16_t)ra[i].x; d[1] = (bf16_t)ra[i].y;
            d[2] = (bf16_t)ra[i].z; d[3] = (bf16_t)ra[i].w;
        }
        #pragma unroll
        for (int i = 0; i < 2; ++i) {
            int f = tid + 256 * i, row = f >> 3, c4 = f & 7;
            bf16_t* d = BsS + row * 48 + c4 * 4;
            d[0] = (bf16_t)rb[i].x; d[1] = (bf16_t)rb[i].y;
            d[2] = (bf16_t)rb[i].z; d[3] = (bf16_t)rb[i].w;
        }
        __syncthreads();

        // ---- issue next K-tile loads (overlap with WMMA below) ----
        if (k0 + 32 < K) {
            #pragma unroll
            for (int i = 0; i < 8; ++i) {
                int f = tid + 256 * i, row = f >> 3, c4 = f & 7;
                ra[i] = *(const float4*)(X + (size_t)(bm + row) * K + (k0 + 32) + c4 * 4);
            }
            #pragma unroll
            for (int i = 0; i < 2; ++i) {
                int f = tid + 256 * i, row = f >> 3, c4 = f & 7;
                rb[i] = *(const float4*)(W + (size_t)(wRowBase + row) * K + (k0 + 32) + c4 * 4);
            }
            if (k0 + 64 < K)   // L2 warm-up two tiles ahead
                __builtin_prefetch(X + (size_t)(bm + (tid >> 1)) * K + (k0 + 64) + (tid & 1) * 16, 0, 3);
        }

        // ---- WMMA: 2 A-frags x 4 B-frags ----
        Frag fa[2];
        #pragma unroll
        for (int mi = 0; mi < 2; ++mi) {
            fa[mi].h[0] = As[wave * 32 + mi * 16 + l16][hsel];
            fa[mi].h[1] = As[wave * 32 + mi * 16 + l16][2 + hsel];
        }
        #pragma unroll
        for (int j = 0; j < 4; ++j) {
            Frag fb;
            fb.h[0] = Bs[j * 16 + l16][hsel];
            fb.h[1] = Bs[j * 16 + l16][2 + hsel];
            #pragma unroll
            for (int mi = 0; mi < 2; ++mi)
                acc[mi][j] = __builtin_amdgcn_wmma_f32_16x16x32_bf16(
                    false, fa[mi].v, false, fb.v, (short)0, acc[mi][j], false, false);
        }
        __syncthreads();
    }

    // ---- epilogue ----
    #pragma unroll
    for (int mi = 0; mi < 2; ++mi)
    #pragma unroll
    for (int j = 0; j < 4; ++j) {
        float bv = bias[wRowBase + j * 16 + l16];
        float* outp = G + (size_t)(bm + wave * 32 + mi * 16 + 8 * hsel) * N5 + bn + j * 16 + l16;
        #pragma unroll
        for (int r = 0; r < 8; ++r)
            outp[(size_t)r * N5] = acc[mi][j][r] + bv;
    }
}

// ---------------------------------------------------------------------------
// Scan: per-channel log-space linear recurrence over S; emits h as bf16.
// ---------------------------------------------------------------------------
__global__ __launch_bounds__(256)
void scan_kernel(const float* __restrict__ G, bf16_t* __restrict__ H,
                 int Bn, int S, int E)
{
    int t = blockIdx.x * blockDim.x + threadIdx.x;
    if (t >= Bn * E) return;
    int b = t / E, e = t - b * E;
    const int N5 = 5 * E;
    const float* g = G + (size_t)b * S * N5 + e;
    bf16_t*      h = H + (size_t)b * S * E + e;

    float m = -0.69314718055994531f;          // log h0 = log 0.5
    for (int s = 0; s < S; ++s) {
        float kc  = g[0];
        float kx  = g[(size_t)E];
        float ktm = g[(size_t)2 * E];
        float kta = g[(size_t)3 * E];
        float ko  = g[(size_t)4 * E];

        float log_a   = log_sigmoid(ktm);
        float log_b   = log_g(kta);
        float log_ctx = log_sigmoid(kc) + log_g(kx);
        float log_o   = log_sigmoid(ko);

        m = logaddexp(log_a + m, log_b);      // log C_t
        float C  = __expf(m);
        float th = 1.0f - 2.0f / (__expf(2.0f * C) + 1.0f);   // tanh(C), C > 0
        float log_tanh = __logf(th + 1e-9f);
        float log_h = logaddexp(log_o + log_tanh, (log_o - ko) + log_ctx);
        *h = (bf16_t)__expf(log_h);

        g += N5; h += E;
    }
}

// ---------------------------------------------------------------------------
// GEMM 2: OUT[M, Nd] = H[M,K](bf16) * Wd[Nd,K]^T    (K = 2048, Nd = 1024)
//   A tile DMA'd via global_load_async_to_lds (double-buffered) when
//   available; B tile register-pipelined with fp32->bf16 conversion.
// ---------------------------------------------------------------------------
__global__ __launch_bounds__(256)
void gemm_down_kernel(const bf16_t* __restrict__ Hbf,
                      const float* __restrict__ Wd,
                      float* __restrict__ OUT,
                      int M, int K, int Nd)
{
    const int bm = blockIdx.x * 256;
    const int bn = blockIdx.y * 64;

    const int tid  = threadIdx.x;
    const int lane = tid & 31;
    const int wave = tid >> 5;
    const int hsel = lane >> 4;
    const int l16  = lane & 15;

    __shared__ v8bf As[2][256][6];
    __shared__ v8bf Bs[64][6];
    bf16_t* BsS = (bf16_t*)&Bs[0][0];

    v8f acc[2][4] = {};
    float4 rb[2];

#if HAVE_ASYNC_LDS
    // ---- issue async DMA of A tile 0 into As[0] ----
    #pragma unroll
    for (int i = 0; i < 4; ++i) {
        int f = tid + 256 * i, row = f >> 2, c8 = f & 3;   // 1024 v8bf chunks
        __builtin_amdgcn_global_load_async_to_lds_b128(
            (__attribute__((address_space(1))) v4i*)(Hbf + (size_t)(bm + row) * K + c8 * 8),
            (__attribute__((address_space(3))) v4i*)&As[0][row][c8], 0, 0);
    }
#else
    v8bf raA[4];
    #pragma unroll
    for (int i = 0; i < 4; ++i) {
        int f = tid + 256 * i, row = f >> 2, c8 = f & 3;
        raA[i] = *(const v8bf*)(Hbf + (size_t)(bm + row) * K + c8 * 8);
    }
#endif
    #pragma unroll
    for (int i = 0; i < 2; ++i) {
        int f = tid + 256 * i, row = f >> 3, c4 = f & 7;
        rb[i] = *(const float4*)(Wd + (size_t)(bn + row) * K + c4 * 4);
    }

    const int T = K / 32;
    for (int kt = 0; kt < T; ++kt) {
        const int k0 = kt * 32, cur = kt & 1;

        // ---- B: convert staged regs -> LDS ----
        #pragma unroll
        for (int i = 0; i < 2; ++i) {
            int f = tid + 256 * i, row = f >> 3, c4 = f & 7;
            bf16_t* d = BsS + row * 48 + c4 * 4;
            d[0] = (bf16_t)rb[i].x; d[1] = (bf16_t)rb[i].y;
            d[2] = (bf16_t)rb[i].z; d[3] = (bf16_t)rb[i].w;
        }
#if HAVE_ASYNC_LDS
        wait_async0();                 // A tile 'cur' DMA complete
#else
        #pragma unroll
        for (int i = 0; i < 4; ++i) {
            int f = tid + 256 * i, row = f >> 2, c8 = f & 3;
            As[cur][row][c8] = raA[i];
        }
#endif
        __syncthreads();

        // ---- issue next tile (overlaps WMMA compute) ----
        if (kt + 1 < T) {
#if HAVE_ASYNC_LDS
            #pragma unroll
            for (int i = 0; i < 4; ++i) {
                int f = tid + 256 * i, row = f >> 2, c8 = f & 3;
                __builtin_amdgcn_global_load_async_to_lds_b128(
                    (__attribute__((address_space(1))) v4i*)(Hbf + (size_t)(bm + row) * K + (k0 + 32) + c8 * 8),
                    (__attribute__((address_space(3))) v4i*)&As[cur ^ 1][row][c8], 0, 0);
            }
#else
            #pragma unroll
            for (int i = 0; i < 4; ++i) {
                int f = tid + 256 * i, row = f >> 2, c8 = f & 3;
                raA[i] = *(const v8bf*)(Hbf + (size_t)(bm + row) * K + (k0 + 32) + c8 * 8);
            }
#endif
            #pragma unroll
            for (int i = 0; i < 2; ++i) {
                int f = tid + 256 * i, row = f >> 3, c4 = f & 7;
                rb[i] = *(const float4*)(Wd + (size_t)(bn + row) * K + (k0 + 32) + c4 * 4);
            }
        }

        // ---- WMMA: 2 A-frags x 4 B-frags ----
        Frag fa[2];
        #pragma unroll
        for (int mi = 0; mi < 2; ++mi) {
            fa[mi].h[0] = As[cur][wave * 32 + mi * 16 + l16][hsel];
            fa[mi].h[1] = As[cur][wave * 32 + mi * 16 + l16][2 + hsel];
        }
        #pragma unroll
        for (int j = 0; j < 4; ++j) {
            Frag fb;
            fb.h[0] = Bs[j * 16 + l16][hsel];
            fb.h[1] = Bs[j * 16 + l16][2 + hsel];
            #pragma unroll
            for (int mi = 0; mi < 2; ++mi)
                acc[mi][j] = __builtin_amdgcn_wmma_f32_16x16x32_bf16(
                    false, fa[mi].v, false, fb.v, (short)0, acc[mi][j], false, false);
        }
        __syncthreads();
    }

    #pragma unroll
    for (int mi = 0; mi < 2; ++mi)
    #pragma unroll
    for (int j = 0; j < 4; ++j) {
        float* outp = OUT + (size_t)(bm + wave * 32 + mi * 16 + 8 * hsel) * Nd + bn + j * 16 + l16;
        #pragma unroll
        for (int r = 0; r < 8; ++r)
            outp[(size_t)r * Nd] = acc[mi][j][r];
    }
}

// ---------------------------------------------------------------------------
extern "C" void kernel_launch(void* const* d_in, const int* in_sizes, int n_in,
                              void* d_out, int out_size, void* d_ws, size_t ws_size,
                              hipStream_t stream)
{
    const float* x      = (const float*)d_in[0];  // [B,S,D]
    const float* W_all  = (const float*)d_in[1];  // [6E,D]
    const float* b_all  = (const float*)d_in[2];  // [6E]
    const float* W_down = (const float*)d_in[3];  // [D,E]

    const int B = 4, S = 2048, D = 1024, E = 2048;
    const int M  = B * S;     // 8192
    const int N5 = 5 * E;     // 10240 (k_e gate is unused -> skipped)

    float*  G = (float*)d_ws;                                   // [M, 5E] fp32
    bf16_t* H = (bf16_t*)((char*)d_ws + (size_t)M * N5 * 4);    // [M, E] bf16

    dim3 g1(M / 256, N5 / 64);   // 32 x 160
    gemm_gates_kernel<<<g1, 256, 0, stream>>>(x, W_all, b_all, G, M, D, E);

    scan_kernel<<<(B * E + 255) / 256, 256, 0, stream>>>(G, H, B, S, E);

    dim3 g2(M / 256, D / 64);    // 32 x 16
    gemm_down_kernel<<<g2, 256, 0, stream>>>(H, W_down, (float*)d_out, M, E, D);
}
